// EncoderLTC_90383291777587
// MI455X (gfx1250) — matrix operand order repaired and checked by reading, not compiled, not run on hardware
//
#include <hip/hip_runtime.h>
#include <hip/hip_bf16.h>

// ---------------------------------------------------------------------------
// LTC encoder for MI455X (gfx1250, wave32, WMMA bf16 16x16x32 f32-accum, TDM)
//
// K0a: pack W_in/W_rec to bf16 pair-dwords, zero h state + barrier counter
// K0b: pack embedding table to bf16 pair-dwords (kills all hot-loop cvt VALU)
// K1 : U[S*B,H] = gather(emb_pk, x) @ W_in + bias  (64x128 block, 8 waves,
//      1 A-frag x 4 B-tiles per wave per k-step)
// K2 : persistent 16-WG scan; W_rec column slice DMA'd into LDS by the
//      Tensor Data Mover (pad-mode, pitch 68 dwords = conflict-free B reads),
//      grid barrier per step, global_prefetch of next step's U slice.
// ---------------------------------------------------------------------------

typedef __attribute__((ext_vector_type(16))) __bf16    v16bf;
typedef __attribute__((ext_vector_type(8)))  float     v8f;
typedef __attribute__((ext_vector_type(4)))  unsigned  u32x4;
typedef __attribute__((ext_vector_type(8)))  int       i32x8;
typedef __attribute__((ext_vector_type(4)))  int       i32x4;

union Frag { unsigned u[8]; v16bf v; };

#define SEQ    512
#define BATCH  64
#define EMBED  512
#define HIDDEN 1024
#define LDS_PITCH 68   /* 64 data dwords + 4 TDM pad dwords per kpair row */

__device__ __forceinline__ unsigned bf16_rne(float f) {
    unsigned u = __float_as_uint(f);
    unsigned r = u + 0x7FFFu + ((u >> 16) & 1u);
    return r >> 16;
}
__device__ __forceinline__ unsigned pack2(float lo, float hi) {
    return bf16_rne(lo) | (bf16_rne(hi) << 16);
}

// ---------------------------------------------------------------------------
// K0a: pack weights, zero h buffers + counter
// ---------------------------------------------------------------------------
__global__ void ltc_pack_weights_kernel(const float* __restrict__ W_in,
                                        const float* __restrict__ W_rec,
                                        unsigned* __restrict__ win_pk,
                                        unsigned* __restrict__ wrec_pk,
                                        float* __restrict__ h32,
                                        unsigned* __restrict__ hpk,
                                        unsigned* __restrict__ counter) {
    const int i = blockIdx.x * blockDim.x + threadIdx.x;   // 512*1024 threads
    if (i < 256 * 1024) {
        int kp = i >> 10, n = i & 1023;
        win_pk[i] = pack2(W_in[(2 * kp) * 1024 + n], W_in[(2 * kp + 1) * 1024 + n]);
    }
    if (i < 512 * 1024) {
        int kp = i >> 10, n = i & 1023;
        wrec_pk[i] = pack2(W_rec[(2 * kp) * 1024 + n], W_rec[(2 * kp + 1) * 1024 + n]);
    }
    if (i < BATCH * HIDDEN) h32[i] = 0.0f;
    if (i < BATCH * HIDDEN / 2) hpk[i] = 0u;
    if (i == 0) *counter = 0u;
}

// ---------------------------------------------------------------------------
// K0b: pack embedding table: emb_pk[token][kpair] = {bf16(2kp), bf16(2kp+1)}
// ---------------------------------------------------------------------------
__global__ void ltc_pack_embed_kernel(const float* __restrict__ embed,
                                      unsigned* __restrict__ emb_pk) {
    const int i = blockIdx.x * blockDim.x + threadIdx.x;   // 32000*256
    if (i >= 32000 * 256) return;
    const float2 e = *(const float2*)(embed + 2 * (long)i);
    emb_pk[i] = pack2(e.x, e.y);
}

// ---------------------------------------------------------------------------
// K1: U = gather(emb_pk, x) @ W_in + bias
// Block = 64 rows x 128 cols. Wave wid: mtile=wid&3, n-group=(wid>>2)*64.
// ---------------------------------------------------------------------------
__global__ void ltc_input_gemm_kernel(const int* __restrict__ x,
                                      const unsigned* __restrict__ emb_pk,
                                      const unsigned* __restrict__ win_pk,
                                      const float* __restrict__ bias,
                                      float* __restrict__ U) {
    const int lane = threadIdx.x & 31;
    const int wid  = threadIdx.x >> 5;
    const int half = lane >> 4;
    const int l16  = lane & 15;

    const int mblk = blockIdx.x >> 3;                 // 512 row-blocks of 64
    const int nblk = blockIdx.x & 7;                  // 8 col-blocks of 128
    const int mtile = mblk * 4 + (wid & 3);
    const int ncb   = nblk * 128 + (wid >> 2) * 64;   // wave's 4 n-tiles base

    const int mrow  = mtile * 16 + l16;               // A-fragment row
    const int token = x[mrow];
    const unsigned* __restrict__ arow = emb_pk + (long)token * (EMBED / 2);

    v8f c[4];
#pragma unroll
    for (int j = 0; j < 4; ++j) {
        const float bv = bias[ncb + j * 16 + l16];
#pragma unroll
        for (int r = 0; r < 8; ++r) c[j][r] = bv;
    }

    for (int kt = 0; kt < EMBED / 32; ++kt) {
        Frag a;
        const int akp = kt * 16 + half * 4;           // A kpairs (kb = half*8)
#pragma unroll
        for (int v = 0; v < 8; ++v)
            a.u[v] = arow[akp + ((v < 4) ? v : v + 4)];
        const int bkp = kt * 16 + half * 4;           // B kpairs (kb = half*8)
#pragma unroll
        for (int j = 0; j < 4; ++j) {
            Frag b;
#pragma unroll
            for (int v = 0; v < 8; ++v)
                b.u[v] = win_pk[(bkp + ((v < 4) ? v : v + 4)) * HIDDEN + ncb + j * 16 + l16];
            c[j] = __builtin_amdgcn_wmma_f32_16x16x32_bf16(false, a.v, false, b.v,
                                                           (short)0, c[j], false, false);
        }
    }
#pragma unroll
    for (int j = 0; j < 4; ++j)
#pragma unroll
        for (int r = 0; r < 8; ++r) {
            const int m = mtile * 16 + half * 8 + r;  // C: VGPR r -> M=r+half*8
            U[(long)m * HIDDEN + ncb + j * 16 + l16] = c[j][r];
        }
}

// ---------------------------------------------------------------------------
// K2: persistent scan. 16 WGs x 256 threads. W_rec slice via TDM into LDS.
// ---------------------------------------------------------------------------
__global__ void ltc_scan_kernel(const float* __restrict__ U,
                                const unsigned* __restrict__ wrec_pk,
                                const float* __restrict__ Avec,
                                const float* __restrict__ tau,
                                float* __restrict__ h32,
                                unsigned* __restrict__ hpk,
                                unsigned* __restrict__ counter,
                                float* __restrict__ out) {
    extern __shared__ unsigned lds[];                 // 512*68 dwords (pad mode)
    const int lane = threadIdx.x & 31;
    const int wid  = threadIdx.x >> 5;
    const int half = lane >> 4;
    const int l16  = lane & 15;
    const int nblocks = gridDim.x;
    const int nslice0 = blockIdx.x * 64;

    // --- TDM: DMA W_rec column slice (512 kpair-rows x 64 dwords, src pitch
    //     1024 dwords) into LDS with 4-dword padding every 64 dwords.
    if (wid == 0) {
        const unsigned long long gaddr =
            (unsigned long long)(const void*)(wrec_pk + nslice0);
        const unsigned ldsaddr = (unsigned)(unsigned long long)(const void*)&lds[0];
        u32x4 g0;
        g0[0] = 1u;                                   // count=1 (valid D#)
        g0[1] = ldsaddr;                              // lds_addr (bytes)
        g0[2] = (unsigned)(gaddr & 0xFFFFFFFFull);    // global_addr[31:0]
        g0[3] = (unsigned)((gaddr >> 32) & 0x1FFFFFFull) | (2u << 30); // +type=2
        i32x8 g1;
        g1[0] = (int)((2u << 16)      |               // data_size = 4B
                      (1u << 20)      |               // pad_enable
                      (5u << 22)      |               // pad_interval = 64 dw
                      (3u << 25));                    // pad_amount   = 4 dw
        g1[1] = (int)(1024u << 16);                   // tensor_dim0 = 1024 (lo16)
        g1[2] = (int)(512u << 16);                    // dim0 hi=0 | tensor_dim1=512
        g1[3] = (int)(64u << 16);                     // dim1 hi=0 | tile_dim0=64
        g1[4] = (int)512;                             // tile_dim1=512 | tile_dim2=0
        g1[5] = (int)1024;                            // tensor_dim0_stride = 1024
        g1[6] = 0;                                    // stride0 hi | stride1 lo
        g1[7] = 0;
        i32x4 gz4 = {0, 0, 0, 0};                     // 2D: groups 2/3 unused
        i32x8 gz8 = {0, 0, 0, 0, 0, 0, 0, 0};        // extra group (unused, 2D)
        __builtin_amdgcn_tensor_load_to_lds(g0, g1, gz4, gz4, gz8, 0);
        __builtin_amdgcn_s_wait_tensorcnt(0);
    }
    __syncthreads();

    const int mtile = wid & 3;
    const int nt0   = (wid >> 2) * 2;
    const int ncol0 = nslice0 + nt0 * 16 + l16;
    const int ncol1 = ncol0 + 16;
    const int nl0   = nt0 * 16 + l16;                 // local cols in LDS row
    const int nl1   = nl0 + 16;
    const int marow = mtile * 16 + l16;
    const float A0 = Avec[ncol0], A1 = Avec[ncol1];
    const float it0 = 1.0f / tau[ncol0], it1 = 1.0f / tau[ncol1];

    for (int t = 0; t < SEQ; ++t) {
        const int rb = t & 1, wb = rb ^ 1;
        const unsigned* __restrict__ hr   = hpk + rb * (BATCH * HIDDEN / 2);
        unsigned* __restrict__       hw   = hpk + wb * (BATCH * HIDDEN / 2);
        const float* __restrict__    hr32 = h32 + rb * (BATCH * HIDDEN);
        float* __restrict__          hw32 = h32 + wb * (BATCH * HIDDEN);
        const float* __restrict__    Ut   = U + (long)t * BATCH * HIDDEN;

        v8f c0, c1;
#pragma unroll
        for (int r = 0; r < 8; ++r) {
            const int m = mtile * 16 + half * 8 + r;
            c0[r] = Ut[m * HIDDEN + ncol0];
            c1[r] = Ut[m * HIDDEN + ncol1];
        }
        // Prefetch next step's U slice into cache while we compute.
        if (t + 1 < SEQ) {
            const float* Un = Ut + BATCH * HIDDEN;
#pragma unroll
            for (int r = 0; r < 8; ++r) {
                const int m = mtile * 16 + half * 8 + r;
                __builtin_prefetch(&Un[m * HIDDEN + ncol0], 0, 1);
            }
        }

        for (int kt = 0; kt < HIDDEN / 32; ++kt) {
            Frag a, b0, b1;
            const int akp = kt * 16 + half * 4;       // A kpairs (kb = half*8)
#pragma unroll
            for (int v = 0; v < 8; ++v)
                a.u[v] = hr[marow * (HIDDEN / 2) + akp + ((v < 4) ? v : v + 4)];
            const int bkp = kt * 16 + half * 4;       // B kpairs (kb = half*8)
#pragma unroll
            for (int v = 0; v < 8; ++v) {
                const int kp = bkp + ((v < 4) ? v : v + 4);
                b0.u[v] = lds[kp * LDS_PITCH + nl0];
                b1.u[v] = lds[kp * LDS_PITCH + nl1];
            }
            c0 = __builtin_amdgcn_wmma_f32_16x16x32_bf16(false, a.v, false, b0.v,
                                                         (short)0, c0, false, false);
            c1 = __builtin_amdgcn_wmma_f32_16x16x32_bf16(false, a.v, false, b1.v,
                                                         (short)0, c1, false, false);
        }

        // Fused tanh + semi-implicit Euler update (DT = 1).
#pragma unroll
        for (int r = 0; r < 8; ++r) {
            const int m = mtile * 16 + half * 8 + r;
            {
                const float f  = tanhf(c0[r]);
                const float hn = (hr32[m * HIDDEN + ncol0] + f * A0) /
                                 (1.0f + it0 + f);
                hw32[m * HIDDEN + ncol0] = hn;
                const unsigned mybf  = bf16_rne(hn);
                const unsigned other = __shfl_xor(mybf, 1, 32);
                if ((lane & 1) == 0)
                    hw[m * (HIDDEN / 2) + (ncol0 >> 1)] = mybf | (other << 16);
            }
            {
                const float f  = tanhf(c1[r]);
                const float hn = (hr32[m * HIDDEN + ncol1] + f * A1) /
                                 (1.0f + it1 + f);
                hw32[m * HIDDEN + ncol1] = hn;
                const unsigned mybf  = bf16_rne(hn);
                const unsigned other = __shfl_xor(mybf, 1, 32);
                if ((lane & 1) == 0)
                    hw[m * (HIDDEN / 2) + (ncol1 >> 1)] = mybf | (other << 16);
            }
        }

        // Grid-wide barrier (monotonic counter; 16 co-resident WGs).
        __threadfence();
        __syncthreads();
        if (threadIdx.x == 0) {
            __hip_atomic_fetch_add(counter, 1u, __ATOMIC_ACQ_REL,
                                   __HIP_MEMORY_SCOPE_AGENT);
            const unsigned target = (unsigned)nblocks * (unsigned)(t + 1);
            while (__hip_atomic_load(counter, __ATOMIC_ACQUIRE,
                                     __HIP_MEMORY_SCOPE_AGENT) < target) {
                __builtin_amdgcn_s_sleep(1);
            }
        }
        __syncthreads();
    }

    // Final h is in buffer 0 (step 511 wrote wb = 0). Copy to d_out [B,H].
    for (int i = blockIdx.x * blockDim.x + threadIdx.x; i < BATCH * HIDDEN;
         i += nblocks * blockDim.x)
        out[i] = h32[i];
}

// ---------------------------------------------------------------------------
// Launch
// ---------------------------------------------------------------------------
extern "C" void kernel_launch(void* const* d_in, const int* in_sizes, int n_in,
                              void* d_out, int out_size, void* d_ws, size_t ws_size,
                              hipStream_t stream) {
    const int*   x     = (const int*)d_in[0];
    const float* embed = (const float*)d_in[1];
    const float* W_in  = (const float*)d_in[2];
    const float* W_rec = (const float*)d_in[3];
    const float* bias  = (const float*)d_in[4];
    const float* tau   = (const float*)d_in[5];
    const float* Avec  = (const float*)d_in[6];
    float* out = (float*)d_out;

    char* ws = (char*)d_ws;
    const size_t MB = 1u << 20;
    unsigned* win_pk  = (unsigned*)(ws);                    // 1 MB
    unsigned* wrec_pk = (unsigned*)(ws + 1 * MB);           // 2 MB
    unsigned* emb_pk  = (unsigned*)(ws + 3 * MB);           // 31.25 MB
    float*    U       = (float*)   (ws + 35 * MB);          // 128 MB
    float*    h32     = (float*)   (ws + 163 * MB);         // 512 KB (x2 bufs)
    unsigned* hpk     = (unsigned*)(ws + 163 * MB + (512u << 10)); // 256 KB
    unsigned* counter = (unsigned*)(ws + 164 * MB);

    ltc_pack_weights_kernel<<<(512 * 1024 + 255) / 256, 256, 0, stream>>>(
        W_in, W_rec, win_pk, wrec_pk, h32, hpk, counter);

    ltc_pack_embed_kernel<<<(32000 * 256 + 255) / 256, 256, 0, stream>>>(
        embed, emb_pk);

    // 512 row-blocks x 8 col-blocks, 256 threads (8 waves) each
    ltc_input_gemm_kernel<<<512 * 8, 256, 0, stream>>>(
        x, emb_pk, win_pk, bias, U);

    // persistent scan: 16 WGs, LDS = 512 * 68 dwords (TDM pad layout)
    ltc_scan_kernel<<<16, 256, 512 * LDS_PITCH * sizeof(unsigned), stream>>>(
        U, wrec_pk, Avec, tau, h32, hpk, counter, out);
}
